// DilatedAttention_4595615006866
// MI455X (gfx1250) — matrix operand order repaired and checked by reading, not compiled, not run on hardware
//
#include <hip/hip_runtime.h>
#include <hip/hip_bf16.h>
#include <math.h>

// ---------------------------------------------------------------------------
// Dilated-attention transformer block for MI455X (gfx1250, wave32, WMMA).
//   x -> LN1 -> QKV (WMMA f16 GEMM, 64x64 LDS weight slab, double-buffered)
//     -> dilated causal attention (online softmax, wave-per-query)
//     -> Wo GEMM + residual (f32) -> LN2 -> W1 GEMM + exact GELU
//     -> W2 GEMM + residual (f32) -> out
// ---------------------------------------------------------------------------

typedef __attribute__((ext_vector_type(16))) _Float16 v16h;
typedef __attribute__((ext_vector_type(8)))  _Float16 v8h;
typedef __attribute__((ext_vector_type(8)))  float    v8f;

#define SEQ_LEN   4096
#define EMBED     512
#define HEADS     8
#define HEAD_DIM  64
#define HIDDEN    2048
#define WINDOW    256
#define DILATION  2

// ---------------------------------------------------------------------------
// f32 -> f16 cast (weights staged once per launch; deterministic)
// ---------------------------------------------------------------------------
__global__ void cast_f32_to_f16(const float* __restrict__ in,
                                _Float16* __restrict__ out, int n) {
  int i = blockIdx.x * blockDim.x + threadIdx.x;
  if (i < n) out[i] = (_Float16)in[i];
}

// ---------------------------------------------------------------------------
// Row LayerNorm over 512 features, f32 in -> f16 out. One block per row.
// ---------------------------------------------------------------------------
__global__ void layernorm_cast_kernel(const float* __restrict__ x,
                                      const float* __restrict__ g,
                                      const float* __restrict__ b,
                                      _Float16* __restrict__ out) {
  const int row = blockIdx.x;
  const int t = threadIdx.x;                    // 0..255
  const float* xr = x + (size_t)row * EMBED;
  float v0 = xr[t];
  float v1 = xr[t + 256];
  float s  = v0 + v1;
  float s2 = v0 * v0 + v1 * v1;
#pragma unroll
  for (int off = 16; off > 0; off >>= 1) {
    s  += __shfl_xor(s,  off, 32);
    s2 += __shfl_xor(s2, off, 32);
  }
  __shared__ float ls[8], ls2[8];
  const int wave = t >> 5, lane = t & 31;
  if (lane == 0) { ls[wave] = s; ls2[wave] = s2; }
  __syncthreads();
  float tot = 0.f, tot2 = 0.f;
#pragma unroll
  for (int i = 0; i < 8; ++i) { tot += ls[i]; tot2 += ls2[i]; }
  const float mu  = tot * (1.f / EMBED);
  const float var = tot2 * (1.f / EMBED) - mu * mu;
  const float rs  = rsqrtf(var + 1e-5f);
  _Float16* orow = out + (size_t)row * EMBED;
  orow[t]       = (_Float16)((v0 - mu) * rs * g[t]       + b[t]);
  orow[t + 256] = (_Float16)((v1 - mu) * rs * g[t + 256] + b[t + 256]);
}

// ---------------------------------------------------------------------------
// WMMA GEMM: out[m,n] = epilogue( sum_k A[m,k] * W[n,k] + bias[n] )
//   A: [M,K] f16 row-major.  W: [N,K] f16 row-major (B = W^T: contiguous K).
// Block: 8 waves, 128(M) x 64(N).  Wave: 16x64 via 4 accumulators.
// 64(N) x 64(K) W slab (8KB) staged in LDS, double-buffered. Per iteration:
//   - issue next slab's global loads (overlap with compute)
//   - issue ALL 16 ds_load_b128 (8 distinct B fragments) up front
//   - 8 back-to-back WMMAs (staggered dscnt waits cover DS latency)
//   - commit next slab, one barrier, rotate buffers.
// MODE: 0 = f16 store, 1 = exact-GELU f16 store, 2 = f32 residual-add store.
// ---------------------------------------------------------------------------
#define LOAD_BF(var, nt, sub)                                                  \
  v16h var = __builtin_shufflevector(                                          \
      *(const v8h*)(&sW[buf][((nt) * 16 + lr) * 64 + (sub) * 32 + 8 * hi]),    \
      *(const v8h*)(&sW[buf][((nt) * 16 + lr) * 64 + (sub) * 32 + 16 + 8 * hi]), \
      0, 1, 2, 3, 4, 5, 6, 7, 8, 9, 10, 11, 12, 13, 14, 15)

#define WMMA_STEP(areg, breg, idx)                                             \
  acc[idx] = __builtin_amdgcn_wmma_f32_16x16x32_f16(                           \
      false, areg, false, breg, (short)0, acc[idx], false, false)

template <int MODE>
__global__ __launch_bounds__(256)
void wmma_gemm_kernel(const _Float16* __restrict__ A,
                      const _Float16* __restrict__ W,
                      const float* __restrict__ bias,
                      const float* __restrict__ res,
                      void* __restrict__ outp,
                      int M, int N, int K) {
  __shared__ _Float16 sW[2][64 * 64];     // [buf][n_local*64 + k_local]

  const int tid  = threadIdx.x;
  const int lane = tid & 31;
  const int wave = tid >> 5;
  const int lr = lane & 15;               // M index (A) / N index (B,C)
  const int hi = lane >> 4;               // half-wave select
  const int m0 = blockIdx.x * 128 + wave * 16;
  const int n0 = blockIdx.y * 64;

  // Cooperative W-slab staging: 64 rows x 128B; 4 threads/row x 32B each.
  const int str = tid >> 2;               // local row 0..63
  const int stc = (tid & 3) * 16;         // half offset 0/16/32/48
  const _Float16* wsrc = W + (size_t)(n0 + str) * K + stc;
  _Float16* sdst0 = &sW[0][str * 64 + stc];
  _Float16* sdst1 = &sW[1][str * 64 + stc];

  const _Float16* arow = A + (size_t)(m0 + lr) * K + 8 * hi;

  v8f acc[4] = {v8f{}, v8f{}, v8f{}, v8f{}};

  // ---- prologue: stage k0 = 0 slab ----
  {
    v16h w0 = *(const v16h*)(wsrc);
    *(v16h*)sdst0 = w0;
  }
  v8h acur[4];
  acur[0] = *(const v8h*)(arow);          // sub-step 0: e=0..7
  acur[1] = *(const v8h*)(arow + 16);     // sub-step 0: e=8..15
  acur[2] = *(const v8h*)(arow + 32);     // sub-step 1
  acur[3] = *(const v8h*)(arow + 48);
  __syncthreads();

  int buf = 0;
  for (int k0 = 0; k0 < K; k0 += 64) {
    const bool more = (k0 + 64 < K);      // scalar: no EXEC divergence

    // Issue next iteration's global loads first (overlap with WMMAs below).
    v16h nw{};
    v8h anext[4] = {{}, {}, {}, {}};
    if (more) {
      nw       = *(const v16h*)(wsrc + k0 + 64);
      anext[0] = *(const v8h*)(arow + k0 + 64);
      anext[1] = *(const v8h*)(arow + k0 + 80);
      anext[2] = *(const v8h*)(arow + k0 + 96);
      anext[3] = *(const v8h*)(arow + k0 + 112);
      __builtin_prefetch(wsrc + k0 + 128, 0, 3);  // global_prefetch_b8
      __builtin_prefetch(arow + k0 + 128, 0, 3);
    }

    // Pack both A fragments: K(e) = 16*(e>>3) + (e&7) + 8*hi (+ 32*sub).
    v16h a0, a1;
#pragma unroll
    for (int e = 0; e < 8; ++e) {
      a0[e]     = acur[0][e];
      a0[e + 8] = acur[1][e];
      a1[e]     = acur[2][e];
      a1[e + 8] = acur[3][e];
    }

    // Issue ALL 8 B-fragment loads (16 ds_load_b128) before any WMMA so the
    // dscnt waits stagger instead of fencing each WMMA at zero.
    LOAD_BF(bf0, 0, 0); LOAD_BF(bf1, 1, 0); LOAD_BF(bf2, 2, 0); LOAD_BF(bf3, 3, 0);
    LOAD_BF(bf4, 0, 1); LOAD_BF(bf5, 1, 1); LOAD_BF(bf6, 2, 1); LOAD_BF(bf7, 3, 1);

    // 8 back-to-back WMMAs; accumulator reuse distance of 4 avoids RAW stalls.
    WMMA_STEP(a0, bf0, 0); WMMA_STEP(a0, bf1, 1);
    WMMA_STEP(a0, bf2, 2); WMMA_STEP(a0, bf3, 3);
    WMMA_STEP(a1, bf4, 0); WMMA_STEP(a1, bf5, 1);
    WMMA_STEP(a1, bf6, 2); WMMA_STEP(a1, bf7, 3);

    // Commit next slab to the alternate buffer, rotate.
    if (more) {
      *(v16h*)(buf ? sdst0 : sdst1) = nw;
#pragma unroll
      for (int e = 0; e < 4; ++e) acur[e] = anext[e];
    }
    __syncthreads();
    buf ^= 1;
  }

  // Epilogue: C layout -> lane holds column (n0+nt*16+lr), rows r+8*hi.
#pragma unroll
  for (int nt = 0; nt < 4; ++nt) {
    const int col = n0 + nt * 16 + lr;
    const float bv = bias[col];
#pragma unroll
    for (int r = 0; r < 8; ++r) {
      const int row = m0 + r + 8 * hi;
      const size_t idx = (size_t)row * N + col;
      float val = acc[nt][r] + bv;
      if (MODE == 0) {
        ((_Float16*)outp)[idx] = (_Float16)val;
      } else if (MODE == 1) {
        float ge = 0.5f * val * (1.f + erff(val * 0.70710678118654752f));
        ((_Float16*)outp)[idx] = (_Float16)ge;
      } else {
        ((float*)outp)[idx] = res[idx] + val;
      }
    }
  }
}

// ---------------------------------------------------------------------------
// Dilated causal attention with online softmax.
// One block per query position i; wave = head; lane owns 2 head dims.
// Keys j = i, i-2, ..., down to max(parity, i-256): <=129 keys.
// ---------------------------------------------------------------------------
__global__ void dilated_attn_kernel(const _Float16* __restrict__ q,
                                    const _Float16* __restrict__ k,
                                    const _Float16* __restrict__ v,
                                    _Float16* __restrict__ out) {
  const int i    = blockIdx.x;          // query position
  const int head = threadIdx.x >> 5;    // 8 waves = 8 heads
  const int lane = threadIdx.x & 31;    // lane owns dims 2*lane, 2*lane+1

  const size_t base = (size_t)i * EMBED + head * HEAD_DIM + 2 * lane;
  const float q0 = (float)q[base];
  const float q1 = (float)q[base + 1];
  const float scale = 0.125f;           // 1/sqrt(64)

  float m = -3.0e38f, l = 0.f, acc0 = 0.f, acc1 = 0.f;

  const int span = (i < WINDOW) ? i : WINDOW;
  const int T = span >> 1;              // j = i - 2t, t = 0..T
  for (int t = 0; t <= T; ++t) {
    const int j = i - DILATION * t;
    const size_t kb = (size_t)j * EMBED + head * HEAD_DIM + 2 * lane;
    float s = q0 * (float)k[kb] + q1 * (float)k[kb + 1];
#pragma unroll
    for (int off = 16; off > 0; off >>= 1) s += __shfl_xor(s, off, 32);
    s *= scale;

    const float mnew = fmaxf(m, s);
    const float c = __expf(m - mnew);   // 0 on first iteration
    const float p = __expf(s - mnew);
    l    = l * c + p;
    acc0 = acc0 * c + p * (float)v[kb];
    acc1 = acc1 * c + p * (float)v[kb + 1];
    m = mnew;
  }

  const float inv = 1.f / l;
  out[base]     = (_Float16)(acc0 * inv);
  out[base + 1] = (_Float16)(acc1 * inv);
}

// ---------------------------------------------------------------------------
// Launch: stage f16 weights, then run the block. All on `stream`.
// ---------------------------------------------------------------------------
extern "C" void kernel_launch(void* const* d_in, const int* in_sizes, int n_in,
                              void* d_out, int out_size, void* d_ws, size_t ws_size,
                              hipStream_t stream) {
  const float* x     = (const float*)d_in[0];
  const float* ln1_g = (const float*)d_in[1];
  const float* ln1_b = (const float*)d_in[2];
  const float* Wq    = (const float*)d_in[3];
  const float* bq    = (const float*)d_in[4];
  const float* Wk    = (const float*)d_in[5];
  const float* bk    = (const float*)d_in[6];
  const float* Wv    = (const float*)d_in[7];
  const float* bv    = (const float*)d_in[8];
  const float* Wo    = (const float*)d_in[9];
  const float* bo    = (const float*)d_in[10];
  const float* ln2_g = (const float*)d_in[11];
  const float* ln2_b = (const float*)d_in[12];
  const float* W1    = (const float*)d_in[13];
  const float* b1    = (const float*)d_in[14];
  const float* W2    = (const float*)d_in[15];
  const float* b2    = (const float*)d_in[16];

  const int L = SEQ_LEN, C = EMBED, H = HIDDEN;

  char* ws = (char*)d_ws;
  size_t off = 0;
  auto wsalloc = [&](size_t bytes) -> void* {
    void* p = ws + off;
    off = (off + bytes + 255) & ~(size_t)255;
    return p;
  };

  _Float16* h16  = (_Float16*)wsalloc((size_t)L * C * 2);
  _Float16* wq16 = (_Float16*)wsalloc((size_t)C * C * 2);
  _Float16* wk16 = (_Float16*)wsalloc((size_t)C * C * 2);
  _Float16* wv16 = (_Float16*)wsalloc((size_t)C * C * 2);
  _Float16* wo16 = (_Float16*)wsalloc((size_t)C * C * 2);
  _Float16* w116 = (_Float16*)wsalloc((size_t)H * C * 2);
  _Float16* w216 = (_Float16*)wsalloc((size_t)C * H * 2);
  _Float16* q16  = (_Float16*)wsalloc((size_t)L * C * 2);
  _Float16* k16  = (_Float16*)wsalloc((size_t)L * C * 2);
  _Float16* v16  = (_Float16*)wsalloc((size_t)L * C * 2);
  float*    x1   = (float*)   wsalloc((size_t)L * C * 4);
  _Float16* g16  = (_Float16*)wsalloc((size_t)L * H * 2);
  _Float16* a16  = h16;   // h16 dead after QKV GEMMs
  _Float16* h2   = q16;   // q16 dead after attention
  (void)ws_size; (void)in_sizes; (void)n_in; (void)out_size;

  // --- stage weights to f16 ---
  {
    const int threads = 256;
    cast_f32_to_f16<<<(C * C + threads - 1) / threads, threads, 0, stream>>>(Wq, wq16, C * C);
    cast_f32_to_f16<<<(C * C + threads - 1) / threads, threads, 0, stream>>>(Wk, wk16, C * C);
    cast_f32_to_f16<<<(C * C + threads - 1) / threads, threads, 0, stream>>>(Wv, wv16, C * C);
    cast_f32_to_f16<<<(C * C + threads - 1) / threads, threads, 0, stream>>>(Wo, wo16, C * C);
    cast_f32_to_f16<<<(H * C + threads - 1) / threads, threads, 0, stream>>>(W1, w116, H * C);
    cast_f32_to_f16<<<(C * H + threads - 1) / threads, threads, 0, stream>>>(W2, w216, C * H);
  }

  // --- LN1 ---
  layernorm_cast_kernel<<<L, 256, 0, stream>>>(x, ln1_g, ln1_b, h16);

  // --- QKV projections (WMMA) ---
  {
    dim3 grid(L / 128, C / 64);
    wmma_gemm_kernel<0><<<grid, 256, 0, stream>>>(h16, wq16, bq, nullptr, q16, L, C, C);
    wmma_gemm_kernel<0><<<grid, 256, 0, stream>>>(h16, wk16, bk, nullptr, k16, L, C, C);
    wmma_gemm_kernel<0><<<grid, 256, 0, stream>>>(h16, wv16, bv, nullptr, v16, L, C, C);
  }

  // --- dilated causal attention ---
  dilated_attn_kernel<<<L, 256, 0, stream>>>(q16, k16, v16, a16);

  // --- output projection + residual (f32) ---
  {
    dim3 grid(L / 128, C / 64);
    wmma_gemm_kernel<2><<<grid, 256, 0, stream>>>(a16, wo16, bo, x, x1, L, C, C);
  }

  // --- LN2 ---
  layernorm_cast_kernel<<<L, 256, 0, stream>>>(x1, ln2_g, ln2_b, h2);

  // --- FFN: W1 + exact GELU, then W2 + residual into d_out ---
  {
    dim3 grid1(L / 128, H / 64);
    wmma_gemm_kernel<1><<<grid1, 256, 0, stream>>>(h2, w116, b1, nullptr, g16, L, H, C);
    dim3 grid2(L / 128, C / 64);
    wmma_gemm_kernel<2><<<grid2, 256, 0, stream>>>(g16, w216, b2, x1, (float*)d_out, L, C, H);
  }
}
